// Packet_MLM_51024211477091
// MI455X (gfx1250) — compile-verified, matrix operand order checked by reading, not run on hardware
//
#include <hip/hip_runtime.h>

// ---------------- problem dims (match reference) ----------------
constexpr int BATCH  = 4;
constexpr int SEQ    = 512;
constexpr int BL     = BATCH * SEQ;      // 2048
constexpr int DMODEL = 1024;
constexpr int DINNER = 2048;
constexpr int NSTATE = 16;
constexpr int DTRANK = 64;
constexpr int XDBLW  = DTRANK + 2 * NSTATE;  // 96 (real)
constexpr int XDBLP  = 128;                  // padded to a full 64x2 tile
constexpr int VOCAB  = 16384;
constexpr int NCLS   = 128;

// ---------------- WMMA types ----------------
typedef __attribute__((ext_vector_type(16))) __bf16         v16bf;
typedef __attribute__((ext_vector_type(8)))  float          v8f;
typedef __attribute__((ext_vector_type(8)))  unsigned short u16x8;

union FragBF { v16bf bf; u16x8 u[2]; };

__device__ __forceinline__ unsigned short f32_to_bf16(float f) {
  unsigned int u = __float_as_uint(f);
  unsigned int lsb = (u >> 16) & 1u;
  u += 0x7fffu + lsb;                 // round-to-nearest-even
  return (unsigned short)(u >> 16);
}

// ---------------- fp32 -> bf16 conversion (weights) ----------------
__global__ void k_cvt_bf16(const float* __restrict__ src,
                           unsigned short* __restrict__ dst, int n) {
  int i = blockIdx.x * blockDim.x + threadIdx.x;
  if (i < n) dst[i] = f32_to_bf16(src[i]);
}

__global__ void k_fill_zero_u16(unsigned short* __restrict__ dst, int n) {
  int i = blockIdx.x * blockDim.x + threadIdx.x;
  if (i < n) dst[i] = 0;
}

// ---------------- embedding gather -> bf16 ----------------
__global__ void k_embed(const int* __restrict__ tok,
                        const float* __restrict__ emb,
                        unsigned short* __restrict__ out) {
  int i = blockIdx.x * blockDim.x + threadIdx.x;   // over BL*DMODEL
  int row = i >> 10;                               // / DMODEL
  int col = i & (DMODEL - 1);
  int t = tok[row];
  out[i] = f32_to_bf16(emb[(size_t)t * DMODEL + col]);
}

// ---------------- generic bf16 WMMA GEMM: C = A(M,K) x W(N,K)^T ----------------
// Block: 128 threads = 4 waves. Block tile 64(M) x 64(N); wave tile 16 x 64.
// A and W fragments both use the symmetric 16-bit WMMA lane layout:
//   row = tile_row + (lane&15); Ks {k..k+7, k+16..k+23} for lanes 0-15,
//   {k+8..k+15, k+24..k+31} for lanes 16-31 (two b128 loads per fragment).
// Requires K % 32 == 0, M % 64 == 0, N % 64 == 0, lda/ldb % 8 == 0 — all
// dims are padded so the inner loop has NO conditionals and NO spills.
template <bool WRITE_BF>
__global__ __launch_bounds__(128, 1)
void k_gemm_bf16(const unsigned short* __restrict__ A,
                 const unsigned short* __restrict__ W,
                 float* __restrict__ C,
                 unsigned short* __restrict__ Cbf,
                 int K, int lda, int ldb, int ldc) {
  const int lane  = threadIdx.x & 31;
  const int wave  = threadIdx.x >> 5;
  const int m0    = blockIdx.y * 64 + wave * 16;
  const int nbase = blockIdx.x * 64;

  const int hi   = lane >> 4;                // 0 or 1
  const int r15  = lane & 15;
  const int koff = hi ? 8 : 0;

  const unsigned short* ap  = A + (size_t)(m0 + r15) * lda + koff;
  const unsigned short* bp0 = W + (size_t)(nbase + r15) * ldb + koff;
  const unsigned short* bp1 = bp0 + (size_t)16 * ldb;
  const unsigned short* bp2 = bp0 + (size_t)32 * ldb;
  const unsigned short* bp3 = bp0 + (size_t)48 * ldb;

  v8f acc0 = {}, acc1 = {}, acc2 = {}, acc3 = {};

  for (int k = 0; k < K; k += 32) {
    FragBF a, b0, b1, b2, b3;
    a.u[0]  = *(const u16x8*)(ap  + k);
    a.u[1]  = *(const u16x8*)(ap  + k + 16);
    b0.u[0] = *(const u16x8*)(bp0 + k);
    b0.u[1] = *(const u16x8*)(bp0 + k + 16);
    b1.u[0] = *(const u16x8*)(bp1 + k);
    b1.u[1] = *(const u16x8*)(bp1 + k + 16);
    b2.u[0] = *(const u16x8*)(bp2 + k);
    b2.u[1] = *(const u16x8*)(bp2 + k + 16);
    b3.u[0] = *(const u16x8*)(bp3 + k);
    b3.u[1] = *(const u16x8*)(bp3 + k + 16);
    // speculative prefetch 8 K tiles (256 elements) ahead (dropped if OOB)
    __builtin_prefetch(ap  + k + 256, 0, 1);
    __builtin_prefetch(bp0 + k + 256, 0, 1);
    __builtin_prefetch(bp2 + k + 256, 0, 1);

    acc0 = __builtin_amdgcn_wmma_f32_16x16x32_bf16(
        false, a.bf, false, b0.bf, (short)0, acc0, false, false);
    acc1 = __builtin_amdgcn_wmma_f32_16x16x32_bf16(
        false, a.bf, false, b1.bf, (short)0, acc1, false, false);
    acc2 = __builtin_amdgcn_wmma_f32_16x16x32_bf16(
        false, a.bf, false, b2.bf, (short)0, acc2, false, false);
    acc3 = __builtin_amdgcn_wmma_f32_16x16x32_bf16(
        false, a.bf, false, b3.bf, (short)0, acc3, false, false);
  }

  // C/D layout: lanes 0-15: N=lane, VGPR r -> M=r ; lanes 16-31: M=r+8
  const int rowb = m0 + 8 * hi;
#pragma unroll
  for (int r = 0; r < 8; ++r) {
    const size_t base = (size_t)(rowb + r) * ldc + r15;
    float v0 = acc0[r], v1 = acc1[r], v2 = acc2[r], v3 = acc3[r];
    C[base + nbase]      = v0;
    C[base + nbase + 16] = v1;
    C[base + nbase + 32] = v2;
    C[base + nbase + 48] = v3;
    if (WRITE_BF) {
      Cbf[base + nbase]      = f32_to_bf16(v0);
      Cbf[base + nbase + 16] = f32_to_bf16(v1);
      Cbf[base + nbase + 32] = f32_to_bf16(v2);
      Cbf[base + nbase + 48] = f32_to_bf16(v3);
    }
  }
}

// ---------------- causal depthwise conv (K=4) + SiLU ----------------
__global__ void k_conv_silu(const float* __restrict__ xz,
                            const float* __restrict__ convw,
                            const float* __restrict__ convb,
                            float* __restrict__ xcf,
                            unsigned short* __restrict__ xcbf) {
  int tid = blockIdx.x * blockDim.x + threadIdx.x;   // 0..8191
  int b = tid >> 11;                                 // / DINNER
  int d = tid & (DINNER - 1);
  float w0 = convw[d * 4 + 0], w1 = convw[d * 4 + 1];
  float w2 = convw[d * 4 + 2], w3 = convw[d * 4 + 3];
  float bias = convb[d];
  float x0 = 0.f, x1 = 0.f, x2 = 0.f, x3 = 0.f;
  for (int l = 0; l < SEQ; ++l) {
    x0 = x1; x1 = x2; x2 = x3;
    x3 = xz[(size_t)(b * SEQ + l) * (2 * DINNER) + d];
    float v = w0 * x0 + w1 * x1 + w2 * x2 + w3 * x3 + bias;
    float s = v / (1.f + __expf(-v));                // SiLU
    size_t o = (size_t)(b * SEQ + l) * DINNER + d;
    xcf[o]  = s;
    xcbf[o] = f32_to_bf16(s);
  }
}

// ---------------- selective scan + D skip + SiLU(z) gate ----------------
// One thread per (b, d) channel: 16 recurrence states in registers,
// per-timestep B/C vectors broadcast through LDS (one batch per block).
__global__ void k_scan(const float* __restrict__ xz,     // z half
                       const float* __restrict__ xcf,    // conv'd x
                       const float* __restrict__ xdbl,   // (BL, XDBLP)
                       const float* __restrict__ dtlin,  // (BL, DINNER) pre-bias
                       const float* __restrict__ dtb,
                       const float* __restrict__ A_log,
                       const float* __restrict__ Dp,
                       unsigned short* __restrict__ yg) {
  __shared__ float sB[NSTATE];
  __shared__ float sC[NSTATE];
  const int b = blockIdx.x >> 3;                   // 8 blocks per batch
  const int d = (blockIdx.x & 7) * 256 + threadIdx.x;

  float Ac[NSTATE], st[NSTATE];
#pragma unroll
  for (int n = 0; n < NSTATE; ++n) {
    Ac[n] = -__expf(A_log[d * NSTATE + n]);
    st[n] = 0.f;
  }
  const float bias  = dtb[d];
  const float dskip = Dp[d];

  for (int l = 0; l < SEQ; ++l) {
    const int row = b * SEQ + l;
    __syncthreads();
    if (threadIdx.x < 2 * NSTATE) {
      float v = xdbl[(size_t)row * XDBLP + DTRANK + threadIdx.x];
      if (threadIdx.x < NSTATE) sB[threadIdx.x] = v;
      else                      sC[threadIdx.x - NSTATE] = v;
    }
    __syncthreads();

    float dl = dtlin[(size_t)row * DINNER + d] + bias;
    float delta = (dl > 20.f) ? dl : log1pf(__expf(dl));   // softplus
    float xv = xcf[(size_t)row * DINNER + d];
    float y = 0.f;
#pragma unroll
    for (int n = 0; n < NSTATE; ++n) {
      float a = __expf(delta * Ac[n]);
      st[n] = a * st[n] + (delta * sB[n]) * xv;
      y += st[n] * sC[n];
    }
    y += xv * dskip;
    float z = xz[(size_t)row * (2 * DINNER) + DINNER + d];
    float g = z / (1.f + __expf(-z));                      // SiLU(z)
    yg[(size_t)row * DINNER + d] = f32_to_bf16(y * g);
  }
}

// ---------------- mean pooling over L ----------------
__global__ void k_pool(const float* __restrict__ hout, float* __restrict__ pooled) {
  int i = blockIdx.x * blockDim.x + threadIdx.x;   // B*DMODEL
  int b = i >> 10;
  int dd = i & (DMODEL - 1);
  float s = 0.f;
  for (int l = 0; l < SEQ; ++l) s += hout[(size_t)(b * SEQ + l) * DMODEL + dd];
  pooled[i] = s * (1.f / SEQ);
}

// ---------------- cls head (tiny) ----------------
__global__ void k_cls(const float* __restrict__ pooled,
                      const float* __restrict__ clsw,
                      float* __restrict__ out) {
  int i = blockIdx.x * blockDim.x + threadIdx.x;
  if (i >= BATCH * NCLS) return;
  int b = i >> 7;
  int c = i & (NCLS - 1);
  float s = 0.f;
  for (int k = 0; k < DMODEL; ++k)
    s += pooled[b * DMODEL + k] * clsw[c * DMODEL + k];
  out[(size_t)BL * VOCAB + i] = s;
}

// ---------------- host side ----------------
extern "C" void kernel_launch(void* const* d_in, const int* in_sizes, int n_in,
                              void* d_out, int out_size, void* d_ws, size_t ws_size,
                              hipStream_t stream) {
  const int*   tokens     = (const int*)  d_in[0];
  const float* embedding  = (const float*)d_in[1];
  const float* in_proj_w  = (const float*)d_in[2];
  const float* conv_w     = (const float*)d_in[3];
  const float* conv_b     = (const float*)d_in[4];
  const float* x_proj_w   = (const float*)d_in[5];
  const float* dt_proj_w  = (const float*)d_in[6];
  const float* dt_proj_b  = (const float*)d_in[7];
  const float* A_log      = (const float*)d_in[8];
  const float* D_param    = (const float*)d_in[9];
  const float* out_proj_w = (const float*)d_in[10];
  const float* recon_w    = (const float*)d_in[11];
  const float* cls_w      = (const float*)d_in[12];
  float* out = (float*)d_out;

  char* ws = (char*)d_ws;
  size_t off = 0;
  auto alloc = [&](size_t bytes) -> void* {
    size_t o = (off + 255) & ~(size_t)255;
    off = o + bytes;
    return (void*)(ws + o);
  };

  unsigned short* emb_bf    = (unsigned short*)alloc((size_t)BL * DMODEL * 2);
  unsigned short* inw_bf    = (unsigned short*)alloc((size_t)2 * DINNER * DMODEL * 2);
  float*          xz        = (float*)         alloc((size_t)BL * 2 * DINNER * 4);
  float*          xconv_f   = (float*)         alloc((size_t)BL * DINNER * 4);
  unsigned short* xconv_bf  = (unsigned short*)alloc((size_t)BL * DINNER * 2);
  unsigned short* xprojw_bf = (unsigned short*)alloc((size_t)XDBLP * DINNER * 2); // padded
  float*          xdbl_f    = (float*)         alloc((size_t)BL * XDBLP * 4);
  unsigned short* xdbl_bf   = (unsigned short*)alloc((size_t)BL * XDBLP * 2);
  unsigned short* dtw_bf    = (unsigned short*)alloc((size_t)DINNER * DTRANK * 2);
  float*          dtlin_f   = (float*)         alloc((size_t)BL * DINNER * 4);
  unsigned short* yg_bf     = (unsigned short*)alloc((size_t)BL * DINNER * 2);
  unsigned short* outw_bf   = (unsigned short*)alloc((size_t)DMODEL * DINNER * 2);
  float*          hout_f    = (float*)         alloc((size_t)BL * DMODEL * 4);
  unsigned short* hout_bf   = (unsigned short*)alloc((size_t)BL * DMODEL * 2);
  unsigned short* reconw_bf = (unsigned short*)alloc((size_t)VOCAB * DMODEL * 2);
  float*          pooled    = (float*)         alloc((size_t)BATCH * DMODEL * 4);

  // 1) one-time-per-call weight conversions fp32 -> bf16
  auto cvt = [&](const float* s, unsigned short* d, int n) {
    k_cvt_bf16<<<(n + 255) / 256, 256, 0, stream>>>(s, d, n);
  };
  cvt(in_proj_w,  inw_bf,    2 * DINNER * DMODEL);
  cvt(x_proj_w,   xprojw_bf, XDBLW * DINNER);
  // zero the 32 pad rows of x_proj_w so padded N-tiles compute zeros
  k_fill_zero_u16<<<((XDBLP - XDBLW) * DINNER + 255) / 256, 256, 0, stream>>>(
      xprojw_bf + (size_t)XDBLW * DINNER, (XDBLP - XDBLW) * DINNER);
  cvt(dt_proj_w,  dtw_bf,    DINNER * DTRANK);
  cvt(out_proj_w, outw_bf,   DMODEL * DINNER);
  cvt(recon_w,    reconw_bf, VOCAB * DMODEL);

  // 2) embedding gather
  k_embed<<<(BL * DMODEL) / 256, 256, 0, stream>>>(tokens, embedding, emb_bf);

  // 3) in_proj: (2048 x 4096 x 1024)
  k_gemm_bf16<false><<<dim3((2 * DINNER) / 64, BL / 64), 128, 0, stream>>>(
      emb_bf, inw_bf, xz, nullptr, DMODEL, DMODEL, DMODEL, 2 * DINNER);

  // 4) causal depthwise conv + SiLU
  k_conv_silu<<<(BATCH * DINNER) / 256, 256, 0, stream>>>(
      xz, conv_w, conv_b, xconv_f, xconv_bf);

  // 5) x_proj: (2048 x 128pad x 2048)
  k_gemm_bf16<true><<<dim3(XDBLP / 64, BL / 64), 128, 0, stream>>>(
      xconv_bf, xprojw_bf, xdbl_f, xdbl_bf, DINNER, DINNER, DINNER, XDBLP);

  // 6) dt_proj: (2048 x 2048 x 64), A = dt columns of x_dbl (lda = XDBLP)
  k_gemm_bf16<false><<<dim3(DINNER / 64, BL / 64), 128, 0, stream>>>(
      xdbl_bf, dtw_bf, dtlin_f, nullptr, DTRANK, XDBLP, DTRANK, DINNER);

  // 7) selective scan + gate
  k_scan<<<(BATCH * DINNER) / 256, 256, 0, stream>>>(
      xz, xconv_f, xdbl_f, dtlin_f, dt_proj_b, A_log, D_param, yg_bf);

  // 8) out_proj: (2048 x 1024 x 2048)
  k_gemm_bf16<true><<<dim3(DMODEL / 64, BL / 64), 128, 0, stream>>>(
      yg_bf, outw_bf, hout_f, hout_bf, DINNER, DINNER, DINNER, DMODEL);

  // 9) recon: (2048 x 16384 x 1024) -> d_out (dominant GEMM)
  k_gemm_bf16<false><<<dim3(VOCAB / 64, BL / 64), 128, 0, stream>>>(
      hout_bf, reconw_bf, out, nullptr, DMODEL, DMODEL, DMODEL, VOCAB);

  // 10) mean pooling + cls head -> tail of d_out
  k_pool<<<(BATCH * DMODEL) / 256, 256, 0, stream>>>(hout_f, pooled);
  k_cls<<<(BATCH * NCLS + 255) / 256, 256, 0, stream>>>(pooled, cls_w, out);
}
